// GNNSpeedup_57964878627467
// MI455X (gfx1250) — matrix operand order repaired and checked by reading, not compiled, not run on hardware
//
#include <hip/hip_runtime.h>
#include <hip/hip_bf16.h>

typedef __attribute__((ext_vector_type(16))) _Float16 v16h;
typedef __attribute__((ext_vector_type(8)))  _Float16 v8h;
typedef __attribute__((ext_vector_type(8)))  float    v8f;

#define NN      20000        // nodes
#define NE      320000       // raw edges
#define ET      (NE + NN)    // edges + self loops
#define HIDDEN  64
#define HEADS   4
#define WIDTH   256          // HIDDEN*HEADS
#define NEG_SLOPE 0.2f
#define LN_EPS  1e-5f
#define MTILES  (NN / 16)    // 1250, exact

// ---------------------------------------------------------------- utilities

__global__ void fill_f32(float* __restrict__ p, float v, int n) {
    int t = blockIdx.x * blockDim.x + threadIdx.x;
    if (t < n) p[t] = v;
}

__device__ __forceinline__ void atomicMaxF(float* addr, float val) {
    // monotonic int encoding trick: signed max for >=0, unsigned min for <0
    if (val >= 0.0f) atomicMax((int*)addr, __float_as_int(val));
    else             atomicMin((unsigned int*)addr, (unsigned int)__float_as_int(val));
}

__device__ __forceinline__ void edge_sd(const int* __restrict__ eidx, int e,
                                        int& s, int& d, bool& valid) {
    if (e < NE) { s = eidx[e]; d = eidx[NE + e]; valid = (s != d); }
    else        { s = d = e - NE; valid = true; }
}

// ------------------------------------------------- weight prepack for WMMA B
// B fragment layout (16-bit, 32x16 KxN per k-tile): lane = n%16 + 16*hi,
// half index j: k = 8*hi + j (j<8), 16 + 8*hi + (j-8) (j>=8). Packed so each lane reads one
// contiguous 32B v16h per (n-tile, k-tile).
__global__ void pack_w(const float* __restrict__ W, int K, int N,
                       _Float16* __restrict__ out) {
    int t = blockIdx.x * blockDim.x + threadIdx.x;
    if (t >= K * N) return;
    int j    = t & 15;
    int lane = (t >> 4) & 31;
    int KT   = K >> 5;
    int kt   = (t >> 9) % KT;
    int nt   = t / (512 * KT);
    int n = nt * 16 + (lane & 15);
    int k = kt * 32 + ((j >> 3) << 4) + ((lane >> 4) << 3) + (j & 7);
    out[t] = (_Float16)W[k * N + n];
}

// ------------------------------------------------------------- input embed
__global__ void input_embed(const float* __restrict__ nf, const float* __restrict__ gf,
                            const float* __restrict__ inW, const float* __restrict__ inb,
                            const float* __restrict__ gW,  const float* __restrict__ gb,
                            const float* __restrict__ bW,  const float* __restrict__ bb,
                            float* __restrict__ x, _Float16* __restrict__ xh) {
    int t = blockIdx.x * blockDim.x + threadIdx.x;
    if (t >= NN * HIDDEN) return;
    int n = t >> 6, j = t & 63;
    float acc = inb[j];
    #pragma unroll
    for (int k = 0; k < 5; ++k) acc += nf[n * 5 + k] * inW[k * HIDDEN + j];
    acc = fmaxf(acc, 0.0f);
    float gamma = gb[j], beta = bb[j];
    #pragma unroll
    for (int k = 0; k < 3; ++k) {
        gamma += gf[k] * gW[k * HIDDEN + j];
        beta  += gf[k] * bW[k * HIDDEN + j];
    }
    float v = acc * (1.0f + gamma) + beta;
    x[t]  = v;
    xh[t] = (_Float16)v;
}

// ------------------------------------------------------------- WMMA GEMM
// A: f16 row-major [16*Mtiles, K=KT*32]. Bpack: prepacked fragments.
// Columns [0,Nsplit) -> out0/bias0, [Nsplit,Ntot) -> out1/bias1.
template<int KT>
__global__ __launch_bounds__(256)
void gemm_wmma(const _Float16* __restrict__ A, int Mtiles,
               const _Float16* __restrict__ Bpack, int Ntot, int Nsplit,
               const float* __restrict__ bias0, const float* __restrict__ bias1,
               float* __restrict__ out0, float* __restrict__ out1,
               int stride0, int stride1, int doRelu) {
    const int K    = KT * 32;
    int wave = threadIdx.x >> 5;
    int lane = threadIdx.x & 31;
    int mt   = blockIdx.x * 8 + wave;
    if (mt >= Mtiles) return;             // wave-uniform: EXEC stays all-1s
    int r  = lane & 15;
    int hi = lane >> 4;
    int row = mt * 16 + r;

    // A fragment per ISA 16-bit A layout:
    //   halfs 0..7  = K (8*hi + 0..7),  halfs 8..15 = K (16 + 8*hi + 0..7)
    union F { v16h v; v8h h[2]; };
    v16h af[KT];
    #pragma unroll
    for (int kt = 0; kt < KT; ++kt) {
        const _Float16* base = A + (size_t)row * K + kt * 32 + hi * 8;
        F u;
        u.h[0] = *(const v8h*)(base);
        u.h[1] = *(const v8h*)(base + 16);
        af[kt] = u.v;
    }

    int NT = Ntot >> 4;
    for (int nt = 0; nt < NT; ++nt) {
        const _Float16* bbase = Bpack + ((size_t)nt * KT * 32 + lane) * 16;
        if (nt + 1 < NT)
            __builtin_prefetch(bbase + (size_t)KT * 512, 0, 1);  // global_prefetch
        v8f acc = {};
        #pragma unroll
        for (int kt = 0; kt < KT; ++kt) {
            v16h bf = *(const v16h*)(bbase + kt * 512);
            acc = __builtin_amdgcn_wmma_f32_16x16x32_f16(
                false, af[kt], false, bf, (short)0, acc, false, false);
        }
        // D layout: VGPR v, lane L: M = v + 8*(L/16), N = L%16
        int col = nt * 16 + r;
        const float* bp = bias0; float* op = out0; int st = stride0; int c = col;
        if (col >= Nsplit) { bp = bias1; op = out1; st = stride1; c = col - Nsplit; }
        float bv = bp[c];
        #pragma unroll
        for (int v = 0; v < 8; ++v) {
            float val = acc[v] + bv;
            if (doRelu) val = fmaxf(val, 0.0f);
            op[(size_t)(mt * 16 + v + 8 * hi) * st + c] = val;
        }
    }
}

// ------------------------------------------------------------- edge phase
__global__ void edge_logits_k(const int* __restrict__ eidx,
                              const float* __restrict__ xl, const float* __restrict__ xr,
                              const float* __restrict__ att,
                              float* __restrict__ logits, float* __restrict__ mbuf) {
    int t = blockIdx.x * blockDim.x + threadIdx.x;
    if (t >= ET * HEADS) return;
    int e = t >> 2, h = t & 3;
    int s, d; bool valid; edge_sd(eidx, e, s, d, valid);
    float lg = -INFINITY;
    if (valid) {
        const float* pl = xl + (size_t)s * WIDTH + h * HIDDEN;
        const float* pr = xr + (size_t)d * WIDTH + h * HIDDEN;
        const float* pa = att + h * HIDDEN;
        float acc = 0.0f;
        #pragma unroll 8
        for (int c = 0; c < HIDDEN; ++c) {
            float v = pl[c] + pr[c];
            v = v >= 0.0f ? v : NEG_SLOPE * v;
            acc += pa[c] * v;
        }
        lg = acc;
        atomicMaxF(mbuf + (size_t)d * HEADS + h, lg);
    }
    logits[t] = lg;
}

__global__ void edge_exp_k(const int* __restrict__ eidx,
                           float* __restrict__ logits,   // in: logits, out: a
                           const float* __restrict__ mbuf,
                           float* __restrict__ denom) {
    int t = blockIdx.x * blockDim.x + threadIdx.x;
    if (t >= ET * HEADS) return;
    int e = t >> 2, h = t & 3;
    int s, d; bool valid; edge_sd(eidx, e, s, d, valid);
    float a = 0.0f;
    if (valid) a = __expf(logits[t] - mbuf[(size_t)d * HEADS + h]);
    logits[t] = a;
    if (a != 0.0f) atomicAdd(denom + (size_t)d * HEADS + h, a);
}

__global__ void edge_scatter_k(const int* __restrict__ eidx,
                               const float* __restrict__ a,
                               const float* __restrict__ denom,
                               const float* __restrict__ xl,
                               float* __restrict__ accum) {
    long long t = (long long)blockIdx.x * blockDim.x + threadIdx.x;
    if (t >= (long long)ET * HIDDEN) return;
    int e = (int)(t >> 6), c = (int)(t & 63);
    int s, d; bool valid; edge_sd(eidx, e, s, d, valid);
    if (!valid) return;
    #pragma unroll
    for (int h = 0; h < HEADS; ++h) {
        float al = a[(size_t)e * HEADS + h];
        if (al == 0.0f) continue;
        float alpha = al / denom[(size_t)d * HEADS + h];
        atomicAdd(accum + (size_t)d * WIDTH + h * HIDDEN + c,
                  alpha * xl[(size_t)s * WIDTH + h * HIDDEN + c]);
    }
}

// --------------------------------------------- bias + LayerNorm + ReLU + res
__global__ __launch_bounds__(256)
void node_update(const float* __restrict__ accum, const float* __restrict__ bias,
                 const float* __restrict__ ln_g,  const float* __restrict__ ln_b,
                 const float* __restrict__ xprev, int residual,
                 float* __restrict__ xout, _Float16* __restrict__ xout_h) {
    int n = blockIdx.x;
    int j = threadIdx.x;            // 0..255
    float v = accum[(size_t)n * WIDTH + j] + bias[j];

    __shared__ float red[8];
    // mean
    float s = v;
    #pragma unroll
    for (int off = 16; off; off >>= 1) s += __shfl_down(s, off, 32);
    if ((j & 31) == 0) red[j >> 5] = s;
    __syncthreads();
    float tot = 0.0f;
    #pragma unroll
    for (int i = 0; i < 8; ++i) tot += red[i];
    float mu = tot * (1.0f / WIDTH);
    __syncthreads();
    // variance
    float dd = v - mu;
    float s2 = dd * dd;
    #pragma unroll
    for (int off = 16; off; off >>= 1) s2 += __shfl_down(s2, off, 32);
    if ((j & 31) == 0) red[j >> 5] = s2;
    __syncthreads();
    float var = 0.0f;
    #pragma unroll
    for (int i = 0; i < 8; ++i) var += red[i];
    var *= (1.0f / WIDTH);

    float y = dd * rsqrtf(var + LN_EPS) * ln_g[j] + ln_b[j];
    y = fmaxf(y, 0.0f);
    if (residual) y += xprev[(size_t)n * WIDTH + j];
    xout[(size_t)n * WIDTH + j]   = y;
    xout_h[(size_t)n * WIDTH + j] = (_Float16)y;
}

// ------------------------------------------------------------- output head 2
__global__ void head2_k(const float* __restrict__ h, const float* __restrict__ W,
                        const float* __restrict__ b, float* __restrict__ out) {
    int t = blockIdx.x * blockDim.x + threadIdx.x;
    if (t >= NN * 3) return;
    int n = t / 3, o = t % 3;
    float acc = b[o];
    #pragma unroll 8
    for (int k = 0; k < HIDDEN; ++k) acc += h[(size_t)n * HIDDEN + k] * W[k * 3 + o];
    out[t] = acc;
}

// ---------------------------------------------------------------------------
extern "C" void kernel_launch(void* const* d_in, const int* in_sizes, int n_in,
                              void* d_out, int out_size, void* d_ws, size_t ws_size,
                              hipStream_t stream) {
    (void)in_sizes; (void)n_in; (void)out_size; (void)ws_size;

    const float* nf   = (const float*)d_in[0];
    const float* gf   = (const float*)d_in[1];
    const int*   eidx = (const int*)d_in[2];
    const float* inW  = (const float*)d_in[3];
    const float* inb  = (const float*)d_in[4];
    const float* gW   = (const float*)d_in[5];
    const float* gb   = (const float*)d_in[6];
    const float* bW   = (const float*)d_in[7];
    const float* bb   = (const float*)d_in[8];
    // layers: 9 + 8*L : Wl, bl, Wr, br, att, bias, ln_g, ln_b
    const float* out1W = (const float*)d_in[41];
    const float* out1b = (const float*)d_in[42];
    const float* out2W = (const float*)d_in[43];
    const float* out2b = (const float*)d_in[44];
    float* out = (float*)d_out;

    // bump allocator on d_ws
    char* ws = (char*)d_ws; size_t off = 0;
    auto alloc = [&](size_t bytes) -> void* {
        off = (off + 255) & ~(size_t)255;
        void* p = ws + off; off += bytes; return p;
    };
    float*    xA    = (float*)   alloc((size_t)NN * WIDTH * 4);
    float*    xB    = (float*)   alloc((size_t)NN * WIDTH * 4);
    _Float16* xh    = (_Float16*)alloc((size_t)NN * WIDTH * 2);
    float*    xl    = (float*)   alloc((size_t)NN * WIDTH * 4);
    float*    xr    = (float*)   alloc((size_t)NN * WIDTH * 4);
    float*    accum = (float*)   alloc((size_t)NN * WIDTH * 4);
    float*    logit = (float*)   alloc((size_t)ET * HEADS * 4);
    float*    mbuf  = (float*)   alloc((size_t)NN * HEADS * 4);
    float*    dbuf  = (float*)   alloc((size_t)NN * HEADS * 4);
    float*    hbuf  = (float*)   alloc((size_t)NN * HIDDEN * 4);
    _Float16* pk[4];
    pk[0] = (_Float16*)alloc((size_t)HIDDEN * 2 * WIDTH * 2);           // K=64,  N=512
    for (int l = 1; l < 4; ++l)
        pk[l] = (_Float16*)alloc((size_t)WIDTH * 2 * WIDTH * 2);        // K=256, N=512
    _Float16* pkO = (_Float16*)alloc((size_t)WIDTH * HIDDEN * 2);       // K=256, N=64

    const int TPB = 256;
    auto cdiv = [](long long a, long long b) { return (int)((a + b - 1) / b); };

    // 1) prepack all WMMA B operands
    for (int l = 0; l < 4; ++l) {
        int K = (l == 0) ? HIDDEN : WIDTH;
        const float* Wl = (const float*)d_in[9 + 8 * l + 0];
        const float* Wr = (const float*)d_in[9 + 8 * l + 2];
        pack_w<<<cdiv((long long)K * WIDTH, TPB), TPB, 0, stream>>>(Wl, K, WIDTH, pk[l]);
        pack_w<<<cdiv((long long)K * WIDTH, TPB), TPB, 0, stream>>>(Wr, K, WIDTH,
                                                                    pk[l] + (size_t)K * WIDTH);
    }
    pack_w<<<cdiv((long long)WIDTH * HIDDEN, TPB), TPB, 0, stream>>>(out1W, WIDTH, HIDDEN, pkO);

    // 2) input embedding + FiLM  (x width 64)
    input_embed<<<cdiv((long long)NN * HIDDEN, TPB), TPB, 0, stream>>>(
        nf, gf, inW, inb, gW, gb, bW, bb, xA, xh);

    // 3) GATv2 layers
    float* xcur = xA;
    float* xnxt = xB;
    for (int l = 0; l < 4; ++l) {
        const float* bl   = (const float*)d_in[9 + 8 * l + 1];
        const float* br   = (const float*)d_in[9 + 8 * l + 3];
        const float* att  = (const float*)d_in[9 + 8 * l + 4];
        const float* bias = (const float*)d_in[9 + 8 * l + 5];
        const float* lng  = (const float*)d_in[9 + 8 * l + 6];
        const float* lnb  = (const float*)d_in[9 + 8 * l + 7];

        // xl = x@Wl+bl ; xr = x@Wr+br  (fused as N=512 WMMA GEMM)
        if (l == 0)
            gemm_wmma<2><<<cdiv(MTILES, 8), 256, 0, stream>>>(
                xh, MTILES, pk[l], 2 * WIDTH, WIDTH, bl, br, xl, xr, WIDTH, WIDTH, 0);
        else
            gemm_wmma<8><<<cdiv(MTILES, 8), 256, 0, stream>>>(
                xh, MTILES, pk[l], 2 * WIDTH, WIDTH, bl, br, xl, xr, WIDTH, WIDTH, 0);

        fill_f32<<<cdiv((long long)NN * HEADS, TPB), TPB, 0, stream>>>(mbuf, -INFINITY, NN * HEADS);
        fill_f32<<<cdiv((long long)NN * HEADS, TPB), TPB, 0, stream>>>(dbuf, 0.0f, NN * HEADS);
        fill_f32<<<cdiv((long long)NN * WIDTH, TPB), TPB, 0, stream>>>(accum, 0.0f, NN * WIDTH);

        edge_logits_k<<<cdiv((long long)ET * HEADS, TPB), TPB, 0, stream>>>(
            eidx, xl, xr, att, logit, mbuf);
        edge_exp_k<<<cdiv((long long)ET * HEADS, TPB), TPB, 0, stream>>>(
            eidx, logit, mbuf, dbuf);
        edge_scatter_k<<<cdiv((long long)ET * HIDDEN, TPB), TPB, 0, stream>>>(
            eidx, logit, dbuf, xl, accum);

        node_update<<<NN, WIDTH, 0, stream>>>(
            accum, bias, lng, lnb, xcur, (l == 0) ? 0 : 1, xnxt, xh);

        float* tmp = xcur; xcur = xnxt; xnxt = tmp;
    }

    // 4) output head: h = relu(x@out1W + b) via WMMA, then small dense
    gemm_wmma<8><<<cdiv(MTILES, 8), 256, 0, stream>>>(
        xh, MTILES, pkO, HIDDEN, HIDDEN, out1b, out1b, hbuf, hbuf, HIDDEN, HIDDEN, 1);
    head2_k<<<cdiv((long long)NN * 3, TPB), TPB, 0, stream>>>(hbuf, out2W, out2b, out);
}
